// MultiHeadAttention_30700426231968
// MI455X (gfx1250) — matrix operand order repaired and checked by reading, not compiled, not run on hardware
//
#include <hip/hip_runtime.h>

#define NSTATE 1024
#define NHEAD  16
#define DHEAD  64
#define BATCH  4
#define SEQ    1500
#define SPAD   1504   // 94 * 16
#define MROWS  6000
#define MPAD   6016   // 94 * 64
#define BH     64     // BATCH * NHEAD

typedef __attribute__((ext_vector_type(16))) _Float16 v16h;
typedef __attribute__((ext_vector_type(8)))  float    v8f;

union Frag {
  v16h     v;
  uint4    q[2];
  _Float16 h[16];
};

__device__ __forceinline__ v8f vzero8() {
  v8f z = {0.f, 0.f, 0.f, 0.f, 0.f, 0.f, 0.f, 0.f};
  return z;
}

__device__ __forceinline__ v8f wmma_f16(v16h a, v16h b, v8f c) {
  // D = A(16x32 f16) * B(32x16 f16) + C(16x16 f32)
  return __builtin_amdgcn_wmma_f32_16x16x32_f16(false, a, false, b, (short)0, c,
                                                false, false);
}

// ---------------- conversion kernels ----------------

__global__ __launch_bounds__(256) void k_cvt_x(const float* __restrict__ x,
                                               _Float16* __restrict__ xb) {
  int idx = blockIdx.x * 256 + threadIdx.x;       // over MPAD*1024
  int row = idx >> 10;
  int col = idx & 1023;
  float v = (row < MROWS) ? x[row * NSTATE + col] : 0.f;
  xb[idx] = (_Float16)v;
}

// Wt[n][k] = W[k][n]  (f32 -> f16, transposed so B-fragments are contiguous)
__global__ __launch_bounds__(256) void k_cvt_wt(const float* __restrict__ w,
                                                _Float16* __restrict__ wt) {
  int idx = blockIdx.x * 256 + threadIdx.x;       // over 1024*1024
  int n  = idx >> 10;
  int kk = idx & 1023;
  wt[idx] = (_Float16)w[kk * NSTATE + n];
}

// ---------------- tiled f16 WMMA GEMM ----------------
// C[m][n] = sum_k A[m][k] * Bt[n][k]; epilogue: (acc + bias[n]) * scale
// mode 0: fp32 out[m*1024+n]
// mode 1: f16 out[b][h][s][d]   (q / k)
// mode 2: f16 out[b][h][d][s]   (v transposed)
__global__ __launch_bounds__(128) void k_gemm(
    const _Float16* __restrict__ A, const _Float16* __restrict__ Bt,
    const float* __restrict__ bias, float scale, int mode,
    _Float16* __restrict__ outh, float* __restrict__ outf) {
  __shared__ __align__(16) _Float16 As[64 * 32];
  __shared__ __align__(16) _Float16 Bs[64 * 32];
  const int tid  = threadIdx.x;
  const int lane = tid & 31;
  const int wid  = tid >> 5;
  const int ln = lane & 15;     // A row / B col / C col within fragment
  const int kh = lane >> 4;     // half-select per ISA 16-bit layouts
  const int m0 = blockIdx.y * 64;
  const int n0 = blockIdx.x * 64;
  const int wm = (wid >> 1) * 32;
  const int wn = (wid & 1) * 32;

  v8f c[2][2];
  c[0][0] = vzero8(); c[0][1] = vzero8();
  c[1][0] = vzero8(); c[1][1] = vzero8();

  const int lrow = tid >> 1;          // 0..63
  const int lk   = (tid & 1) * 16;    // 0 / 16

  for (int k0 = 0; k0 < NSTATE; k0 += 32) {
    // cooperative global -> LDS (each thread moves 32B of A and 32B of B)
    const uint4* sa = (const uint4*)(A + (size_t)(m0 + lrow) * NSTATE + k0 + lk);
    uint4* da = (uint4*)(As + lrow * 32 + lk);
    da[0] = sa[0]; da[1] = sa[1];
    const uint4* sb = (const uint4*)(Bt + (size_t)(n0 + lrow) * NSTATE + k0 + lk);
    uint4* db = (uint4*)(Bs + lrow * 32 + lk);
    db[0] = sb[0]; db[1] = sb[1];
    __syncthreads();

    Frag a[2], b[2];
#pragma unroll
    for (int mi = 0; mi < 2; ++mi) {
      // A-fragment: elems 0..7 -> K = kh*8+i ; elems 8..15 -> K = 16+kh*8+i
      const _Float16* p = As + (wm + mi * 16 + ln) * 32;
      a[mi].q[0] = *(const uint4*)(p + kh * 8);
      a[mi].q[1] = *(const uint4*)(p + 16 + kh * 8);
    }
#pragma unroll
    for (int ni = 0; ni < 2; ++ni) {
      // B-fragment: elems 0..7 -> K = kh*8+i ; elems 8..15 -> K = 16+kh*8+i
      const _Float16* p = Bs + (wn + ni * 16 + ln) * 32;
      b[ni].q[0] = *(const uint4*)(p + kh * 8);
      b[ni].q[1] = *(const uint4*)(p + 16 + kh * 8);
    }
#pragma unroll
    for (int mi = 0; mi < 2; ++mi)
#pragma unroll
      for (int ni = 0; ni < 2; ++ni)
        c[mi][ni] = wmma_f16(a[mi].v, b[ni].v, c[mi][ni]);
    __syncthreads();
  }

  // epilogue: C layout -> lane holds col n, rows kh*8 + r
#pragma unroll
  for (int mi = 0; mi < 2; ++mi) {
#pragma unroll
    for (int ni = 0; ni < 2; ++ni) {
      int n = n0 + wn + ni * 16 + ln;
      float bv = (bias != nullptr) ? bias[n] : 0.f;
#pragma unroll
      for (int r = 0; r < 8; ++r) {
        int m = m0 + wm + mi * 16 + kh * 8 + r;
        if (m >= MROWS) continue;
        float val = (c[mi][ni][r] + bv) * scale;
        if (mode == 0) {
          outf[(size_t)m * NSTATE + n] = val;
        } else {
          int bb = m / SEQ;
          int s  = m - bb * SEQ;
          int h  = n >> 6;
          int d  = n & 63;
          if (mode == 1)
            outh[(((size_t)(bb * NHEAD + h) * SPAD) + s) * DHEAD + d] = (_Float16)val;
          else
            outh[(((size_t)(bb * NHEAD + h) * DHEAD) + d) * SPAD + s] = (_Float16)val;
        }
      }
    }
  }
}

// ---------------- flash attention (one wave = 16 q rows of one head) ----------------
__global__ __launch_bounds__(32) void k_attn(
    const _Float16* __restrict__ qb, const _Float16* __restrict__ kb,
    const _Float16* __restrict__ vt, _Float16* __restrict__ ob) {
  __shared__ __align__(16) _Float16 Ps[32 * 16];  // [key col][q row]
  const int lane = threadIdx.x;
  const int ln = lane & 15;
  const int kh = lane >> 4;
  const int bh = blockIdx.x;   // 0..63
  const int qt = blockIdx.y;   // 0..93

  const _Float16* qp = qb + ((size_t)bh * SPAD + qt * 16) * DHEAD;
  const _Float16* kp = kb + (size_t)bh * SPAD * DHEAD;
  const _Float16* vp = vt + (size_t)bh * DHEAD * SPAD;  // [d][s]

  Frag qa[2];  // q tile 16x64 as two 16x32 A-fragments
#pragma unroll
  for (int dd = 0; dd < 2; ++dd) {
    const _Float16* p = qp + ln * DHEAD + dd * 32;
    qa[dd].q[0] = *(const uint4*)(p + kh * 8);
    qa[dd].q[1] = *(const uint4*)(p + 16 + kh * 8);
  }

  v8f o[4];
  o[0] = vzero8(); o[1] = vzero8(); o[2] = vzero8(); o[3] = vzero8();
  float mrow[8], lrowv[8];
#pragma unroll
  for (int r = 0; r < 8; ++r) { mrow[r] = -3.0e38f; lrowv[r] = 0.f; }

  for (int t2 = 0; t2 < SPAD / 32; ++t2) {  // 47 key-tile pairs
    const int key0 = t2 * 32;
    v8f c0 = vzero8(), c1 = vzero8();
#pragma unroll
    for (int dd = 0; dd < 2; ++dd) {
      Frag b0, b1;  // B = K^T [d][key]; lane col = key, rows = kh*8+i / 16+kh*8+i
      const _Float16* p0 = kp + (size_t)(key0 + ln) * DHEAD + dd * 32;
      b0.q[0] = *(const uint4*)(p0 + kh * 8);
      b0.q[1] = *(const uint4*)(p0 + 16 + kh * 8);
      const _Float16* p1 = kp + (size_t)(key0 + 16 + ln) * DHEAD + dd * 32;
      b1.q[0] = *(const uint4*)(p1 + kh * 8);
      b1.q[1] = *(const uint4*)(p1 + 16 + kh * 8);
      c0 = wmma_f16(qa[dd].v, b0.v, c0);
      c1 = wmma_f16(qa[dd].v, b1.v, c1);
    }
    // mask padded keys (index-based)
    if (key0 + ln >= SEQ) {
#pragma unroll
      for (int r = 0; r < 8; ++r) c0[r] = -3.0e38f;
    }
    if (key0 + 16 + ln >= SEQ) {
#pragma unroll
      for (int r = 0; r < 8; ++r) c1[r] = -3.0e38f;
    }
    // online softmax per row (rows kh*8+r, cols across 16 lanes of the group)
#pragma unroll
    for (int r = 0; r < 8; ++r) {
      float tm = fmaxf(c0[r], c1[r]);
      tm = fmaxf(tm, __shfl_xor(tm, 1, 32));
      tm = fmaxf(tm, __shfl_xor(tm, 2, 32));
      tm = fmaxf(tm, __shfl_xor(tm, 4, 32));
      tm = fmaxf(tm, __shfl_xor(tm, 8, 32));
      float mn = fmaxf(mrow[r], tm);
      float al = __expf(mrow[r] - mn);
      float p0 = __expf(c0[r] - mn);
      float p1 = __expf(c1[r] - mn);
      float ps = p0 + p1;
      ps += __shfl_xor(ps, 1, 32);
      ps += __shfl_xor(ps, 2, 32);
      ps += __shfl_xor(ps, 4, 32);
      ps += __shfl_xor(ps, 8, 32);
      lrowv[r] = lrowv[r] * al + ps;
      mrow[r] = mn;
      c0[r] = p0;
      c1[r] = p1;
#pragma unroll
      for (int g = 0; g < 4; ++g) o[g][r] *= al;
    }
    // stage P (C layout) into LDS column-major, read back as A-fragment
#pragma unroll
    for (int r = 0; r < 8; ++r) {
      Ps[ln * 16 + kh * 8 + r]        = (_Float16)c0[r];
      Ps[(16 + ln) * 16 + kh * 8 + r] = (_Float16)c1[r];
    }
    __syncthreads();
    Frag pf;
#pragma unroll
    for (int i = 0; i < 8; ++i) {
      pf.h[i]     = Ps[(kh * 8 + i) * 16 + ln];
      pf.h[8 + i] = Ps[(16 + kh * 8 + i) * 16 + ln];
    }
    __syncthreads();
    // o += P(16x32) @ V(32x64); V^T layout makes B-fragments contiguous
#pragma unroll
    for (int g = 0; g < 4; ++g) {
      Frag vb;
      const _Float16* pv = vp + (size_t)(g * 16 + ln) * SPAD + key0;
      vb.q[0] = *(const uint4*)(pv + kh * 8);
      vb.q[1] = *(const uint4*)(pv + 16 + kh * 8);
      o[g] = wmma_f16(pf.v, vb.v, o[g]);
    }
  }

  // normalize + store into [b][s][h*64+d] f16 for the output projection
  const int h  = bh & (NHEAD - 1);
  const int bb = bh >> 4;
#pragma unroll
  for (int r = 0; r < 8; ++r) {
    int s = qt * 16 + kh * 8 + r;
    if (s >= SEQ) continue;
    float inv = 1.0f / lrowv[r];
#pragma unroll
    for (int g = 0; g < 4; ++g) {
      int col = h * DHEAD + g * 16 + ln;
      ob[((size_t)(bb * SEQ + s)) * NSTATE + col] = (_Float16)(o[g][r] * inv);
    }
  }
}

// ---------------- host launcher ----------------
extern "C" void kernel_launch(void* const* d_in, const int* in_sizes, int n_in,
                              void* d_out, int out_size, void* d_ws, size_t ws_size,
                              hipStream_t stream) {
  (void)in_sizes; (void)n_in; (void)out_size; (void)ws_size;
  const float* x  = (const float*)d_in[0];
  const float* Wq = (const float*)d_in[1];
  const float* bq = (const float*)d_in[2];
  const float* Wk = (const float*)d_in[3];
  const float* Wv = (const float*)d_in[4];
  const float* bv = (const float*)d_in[5];
  const float* Wo = (const float*)d_in[6];
  const float* bo = (const float*)d_in[7];
  float* out = (float*)d_out;

  char* ws = (char*)d_ws;
  const size_t SZ_XB  = (size_t)MPAD * NSTATE * sizeof(_Float16);   // 12.3 MB
  const size_t SZ_WT  = (size_t)NSTATE * NSTATE * sizeof(_Float16); //  2.1 MB
  const size_t SZ_QKV = (size_t)BH * SPAD * DHEAD * sizeof(_Float16);
  _Float16* xb   = (_Float16*)(ws);
  _Float16* wtq  = (_Float16*)(ws + SZ_XB);
  _Float16* wtk  = (_Float16*)(ws + SZ_XB + 1 * SZ_WT);
  _Float16* wtv  = (_Float16*)(ws + SZ_XB + 2 * SZ_WT);
  _Float16* wto  = (_Float16*)(ws + SZ_XB + 3 * SZ_WT);
  _Float16* qbuf = (_Float16*)(ws + SZ_XB + 4 * SZ_WT);
  _Float16* kbuf = (_Float16*)(ws + SZ_XB + 4 * SZ_WT + 1 * SZ_QKV);
  _Float16* vtb  = (_Float16*)(ws + SZ_XB + 4 * SZ_WT + 2 * SZ_QKV);
  _Float16* ob   = (_Float16*)(ws + SZ_XB + 4 * SZ_WT + 3 * SZ_QKV);
  // total ws use: ~70 MB

  const float scale = 0.35355339059327373f;  // 64^-0.25, applied to q and k

  k_cvt_x<<<(MPAD * NSTATE) / 256, 256, 0, stream>>>(x, xb);
  k_cvt_wt<<<(NSTATE * NSTATE) / 256, 256, 0, stream>>>(Wq, wtq);
  k_cvt_wt<<<(NSTATE * NSTATE) / 256, 256, 0, stream>>>(Wk, wtk);
  k_cvt_wt<<<(NSTATE * NSTATE) / 256, 256, 0, stream>>>(Wv, wtv);
  k_cvt_wt<<<(NSTATE * NSTATE) / 256, 256, 0, stream>>>(Wo, wto);

  dim3 gg(NSTATE / 64, MPAD / 64);
  k_gemm<<<gg, 128, 0, stream>>>(xb, wtq, bq,      scale, 1, qbuf, nullptr);
  k_gemm<<<gg, 128, 0, stream>>>(xb, wtk, nullptr, scale, 1, kbuf, nullptr);
  k_gemm<<<gg, 128, 0, stream>>>(xb, wtv, bv,      1.0f,  2, vtb,  nullptr);

  dim3 ga(BH, SPAD / 16);
  k_attn<<<ga, 32, 0, stream>>>(qbuf, kbuf, vtb, ob);

  k_gemm<<<gg, 128, 0, stream>>>(ob, wto, bo, 1.0f, 0, nullptr, out);
}
